// StructuralTransformerLayer_16741782520605
// MI455X (gfx1250) — matrix operand order repaired and checked by reading, not compiled, not run on hardware
//
#include <hip/hip_runtime.h>
#include <hip/hip_bf16.h>
#include <cstdint>

typedef _Float16 f16;
typedef __attribute__((ext_vector_type(16))) _Float16 v16h;
typedef __attribute__((ext_vector_type(8)))  float    v8f;

#define DD   768
#define LL   1024
#define BBS  4
#define HHN  8
#define HD   96
#define DFF  2048
#define NTOK 4096   // B*L

// ---- CDNA5 async global->LDS staging (guarded: fall back to sync copies) ----
#if defined(__has_builtin)
#if __has_builtin(__builtin_amdgcn_global_load_async_to_lds_b128) && \
    __has_builtin(__builtin_amdgcn_s_wait_asynccnt)
#define USE_ASYNC_LDS 1
#endif
#endif

#if defined(USE_ASYNC_LDS)
typedef int i4v __attribute__((ext_vector_type(4)));
typedef __attribute__((address_space(1))) i4v* gp128_t;  // global int4*
typedef __attribute__((address_space(3))) i4v* lp128_t;  // LDS int4* (32-bit)
static __device__ __forceinline__ void async_copy_b128(const void* g, void* l) {
  // generic->as1: same numeric value; generic->as3: LDS offset = addr[31:0]
  __builtin_amdgcn_global_load_async_to_lds_b128(
      (gp128_t)(uintptr_t)g, (lp128_t)(unsigned int)(uintptr_t)l, 0, 0);
}
static __device__ __forceinline__ void async_wait0() {
  __builtin_amdgcn_s_wait_asynccnt(0);
}
#endif

static __device__ __forceinline__ v8f wmma16x16x32(v16h a, v16h b, v8f c) {
  return __builtin_amdgcn_wmma_f32_16x16x32_f16(false, a, false, b, (short)0, c,
                                                false, false);
}

static __device__ __forceinline__ v8f vzero8() {
  v8f z;
#pragma unroll
  for (int i = 0; i < 8; ++i) z[i] = 0.0f;
  return z;
}

__global__ void cvt_f32_f16(const float* __restrict__ in, f16* __restrict__ out,
                            int n) {
  for (int i = blockIdx.x * blockDim.x + threadIdx.x; i < n;
       i += gridDim.x * blockDim.x)
    out[i] = (f16)in[i];
}

// ---------------------------------------------------------------------------
// WMMA GEMM: C[M,N] = A[M,K](f16,row-major) * Bw[N,K]^T + bias
// Block tile 128(M) x 128(N), 8 waves (2 in M x 4 in N), wave = 64x32
// = 4x2 WMMA tiles, so 6 fragment loads feed 8 WMMAs per K-step.
// EPI: 0 = scatter to head-major q/k/v (f16);  1 = +residual -> f32;
//      2 = exact GELU -> f16
// ---------------------------------------------------------------------------
template <int EPI>
__global__ __launch_bounds__(256) void gemm_wmma(
    const f16* __restrict__ A, int lda, const f16* __restrict__ Bw,
    const float* __restrict__ bias, int N, int K, float* __restrict__ outF,
    f16* __restrict__ outH, const float* __restrict__ res) {
  __shared__ __align__(16) f16 As[128][40];  // 80B row stride: aligned, bank-safe
  __shared__ __align__(16) f16 Bs[128][40];

  const int t = threadIdx.x;
  const int wave = t >> 5;
  const int lane = t & 31;
  const int laneM = lane & 15;
  const int half = lane >> 4;
  const int wm = (wave & 1) * 64;
  const int wn = (wave >> 1) * 32;
  const int m0 = blockIdx.y * 128;
  const int n0 = blockIdx.x * 128;

  v8f acc[4][2];
#pragma unroll
  for (int i = 0; i < 4; ++i)
#pragma unroll
    for (int j = 0; j < 2; ++j) acc[i][j] = vzero8();

  const int srow = t >> 1, sc16 = (t & 1) * 16;  // 2 threads/row * 128 rows

  for (int k0 = 0; k0 < K; k0 += 32) {
    const f16* ga = A + (size_t)(m0 + srow) * lda + k0 + sc16;
    const f16* gb = Bw + (size_t)(n0 + srow) * K + k0 + sc16;
#if defined(USE_ASYNC_LDS)
    async_copy_b128(ga, &As[srow][sc16]);
    async_copy_b128(ga + 8, &As[srow][sc16 + 8]);
    async_copy_b128(gb, &Bs[srow][sc16]);
    async_copy_b128(gb + 8, &Bs[srow][sc16 + 8]);
    async_wait0();
#else
    *(float4*)(&As[srow][sc16]) = *(const float4*)(ga);
    *(float4*)(&As[srow][sc16 + 8]) = *(const float4*)(ga + 8);
    *(float4*)(&Bs[srow][sc16]) = *(const float4*)(gb);
    *(float4*)(&Bs[srow][sc16 + 8]) = *(const float4*)(gb + 8);
    if (k0 + 32 < K) {  // CDNA5 global_prefetch_b8 for next tiles
      __builtin_prefetch(ga + 32, 0, 1);
      __builtin_prefetch(gb + 32, 0, 1);
    }
#endif
    __syncthreads();

    v16h af[4], bf[2];
#pragma unroll
    for (int mi = 0; mi < 4; ++mi) {
#pragma unroll
      for (int i = 0; i < 8; ++i) {  // A layout: half0 K=0..7,16..23; half1 +8
        const int kk = (i < 4 ? 2 * i : 16 + 2 * (i - 4)) + half * 8;
        af[mi][2 * i] = As[wm + mi * 16 + laneM][kk];
        af[mi][2 * i + 1] = As[wm + mi * 16 + laneM][kk + 1];
      }
    }
#pragma unroll
    for (int nj = 0; nj < 2; ++nj) {
#pragma unroll
      for (int v = 0; v < 8; ++v) {  // B layout: half0 K=0..7,16..23; half1 +8
        const int kk = (v < 4 ? 2 * v : 16 + 2 * (v - 4)) + half * 8;
        bf[nj][2 * v] = Bs[wn + nj * 16 + laneM][kk];
        bf[nj][2 * v + 1] = Bs[wn + nj * 16 + laneM][kk + 1];
      }
    }
#pragma unroll
    for (int mi = 0; mi < 4; ++mi)
#pragma unroll
      for (int nj = 0; nj < 2; ++nj)
        acc[mi][nj] = wmma16x16x32(af[mi], bf[nj], acc[mi][nj]);
    __syncthreads();
  }

#pragma unroll
  for (int mi = 0; mi < 4; ++mi) {
#pragma unroll
    for (int nj = 0; nj < 2; ++nj) {
      const int n = n0 + wn + nj * 16 + laneM;
      const float bv = bias[n];
#pragma unroll
      for (int r = 0; r < 8; ++r) {  // C layout: row = r + half*8
        const int m = m0 + wm + mi * 16 + half * 8 + r;
        const float val = acc[mi][nj][r] + bv;
        if (EPI == 0) {
          const int which = n / DD;
          const int rem = n - which * DD;
          const int hh = rem / HD;
          const int d = rem - hh * HD;
          const int bb = m >> 10;
          const int l = m & 1023;
          outH[(size_t)which * ((size_t)BBS * HHN * LL * HD) +
               ((size_t)((bb * HHN + hh) * LL + l)) * HD + d] = (f16)val;
        } else if (EPI == 1) {
          const size_t idx = (size_t)m * N + n;
          outF[idx] = val + res[idx];
        } else {
          const float ge = 0.5f * val * (1.0f + erff(val * 0.70710678118f));
          outH[(size_t)m * N + n] = (f16)ge;
        }
      }
    }
  }
}

// ---------------------------------------------------------------------------
// Flash attention: block = one (b,h), 8 waves x 16 queries = 128 queries.
// K/V streamed in 32-key tiles through LDS shared by all waves.
// ---------------------------------------------------------------------------
__global__ __launch_bounds__(256) void flash_attn(const f16* __restrict__ qkvh,
                                                  const float* __restrict__ abias,
                                                  f16* __restrict__ attnh) {
  __shared__ __align__(16) f16 Ks[32][104];  // 208B rows: 16B-aligned, bank-safe
  __shared__ __align__(16) f16 Vs[32][104];
  __shared__ __align__(16) f16 Ps[8][16][36]; // per-wave P transpose scratch

  const int t = threadIdx.x;
  const int w = t >> 5, lane = t & 31, laneM = lane & 15, half = lane >> 4;
  const int bh = blockIdx.y;
  const int b = bh >> 3, h = bh & 7;
  const int qbase = blockIdx.x * 128 + w * 16;
  const size_t S1 = (size_t)BBS * HHN * LL * HD;
  const f16* qp = qkvh + (size_t)bh * LL * HD;
  const f16* kp = qkvh + S1 + (size_t)bh * LL * HD;
  const f16* vp = qkvh + 2 * S1 + (size_t)bh * LL * HD;
  const float scale = 0.1020620726f;  // 1/sqrt(96)

  v16h aq[3];  // Q fragments, K-dim = 96 = 3 chunks of 32; scale folded in
#pragma unroll
  for (int c = 0; c < 3; ++c)
#pragma unroll
    for (int i = 0; i < 8; ++i) {
      const int kk = c * 32 + half * 8 + (i < 4 ? 2 * i : 16 + 2 * (i - 4));
      const f16* qr = qp + (size_t)(qbase + laneM) * HD + kk;
      aq[c][2 * i] = (f16)((float)qr[0] * scale);
      aq[c][2 * i + 1] = (f16)((float)qr[1] * scale);
    }

  float mrun[8], lrun[8];
#pragma unroll
  for (int r = 0; r < 8; ++r) { mrun[r] = -3.0e38f; lrun[r] = 0.0f; }
  v8f o[6];
#pragma unroll
  for (int d = 0; d < 6; ++d) o[d] = vzero8();

  for (int kb = 0; kb < 32; ++kb) {
#if defined(USE_ASYNC_LDS)
    if (t < 192) {                       // 32 rows * 6 x 16B chunks
      const int row = t / 6, c8 = (t % 6) * 8;
      async_copy_b128(kp + (size_t)(kb * 32 + row) * HD + c8, &Ks[row][c8]);
      async_copy_b128(vp + (size_t)(kb * 32 + row) * HD + c8, &Vs[row][c8]);
    }
    async_wait0();
#else
    {
      const int lrow = t >> 3, lc = (t & 7) * 6;  // 8 thr/row, 6 dwords each
      const unsigned int* gk =
          (const unsigned int*)(kp + (size_t)(kb * 32 + lrow) * HD);
      const unsigned int* gv =
          (const unsigned int*)(vp + (size_t)(kb * 32 + lrow) * HD);
      unsigned int* sk = (unsigned int*)(&Ks[lrow][0]);
      unsigned int* sv = (unsigned int*)(&Vs[lrow][0]);
#pragma unroll
      for (int j = 0; j < 6; ++j) {
        sk[lc + j] = gk[lc + j];
        sv[lc + j] = gv[lc + j];
      }
    }
#endif
    __syncthreads();

    // S = Q*K^T for 2 key sub-tiles of 16
    v8f s0 = vzero8(), s1 = vzero8();
#pragma unroll
    for (int c = 0; c < 3; ++c) {
      v16h b0, b1;
#pragma unroll
      for (int v = 0; v < 8; ++v) {
        const int dd = c * 32 + half * 8 + (v < 4 ? 2 * v : 16 + 2 * (v - 4));
        b0[2 * v] = Ks[laneM][dd];
        b0[2 * v + 1] = Ks[laneM][dd + 1];
        b1[2 * v] = Ks[16 + laneM][dd];
        b1[2 * v + 1] = Ks[16 + laneM][dd + 1];
      }
      s0 = wmma16x16x32(aq[c], b0, s0);
      s1 = wmma16x16x32(aq[c], b1, s1);
    }

    // bias add + online softmax (stats per row, rows = r + half*8 in C layout)
#pragma unroll
    for (int r = 0; r < 8; ++r) {
      const int row = half * 8 + r;
      const float* bp =
          abias + ((size_t)b * LL + (qbase + row)) * LL + kb * 32 + laneM;
      const float v0 = s0[r] + bp[0];
      const float v1 = s1[r] + bp[16];
      float mx = fmaxf(v0, v1);
      mx = fmaxf(mx, __shfl_xor(mx, 1));
      mx = fmaxf(mx, __shfl_xor(mx, 2));
      mx = fmaxf(mx, __shfl_xor(mx, 4));
      mx = fmaxf(mx, __shfl_xor(mx, 8));  // reduce within 16-lane half
      const float mnew = fmaxf(mrun[r], mx);
      const float fact = expf(mrun[r] - mnew);
      const float p0 = expf(v0 - mnew);
      const float p1 = expf(v1 - mnew);
      float rs = p0 + p1;
      rs += __shfl_xor(rs, 1);
      rs += __shfl_xor(rs, 2);
      rs += __shfl_xor(rs, 4);
      rs += __shfl_xor(rs, 8);
      lrun[r] = lrun[r] * fact + rs;
      mrun[r] = mnew;
#pragma unroll
      for (int dt = 0; dt < 6; ++dt) o[dt][r] *= fact;
      Ps[w][row][laneM] = (f16)p0;        // C-layout -> LDS (per-wave region)
      Ps[w][row][16 + laneM] = (f16)p1;
    }
    __builtin_amdgcn_wave_barrier();  // keep ds store->load program order

    v16h pa;  // read P back in A-fragment layout
#pragma unroll
    for (int i = 0; i < 8; ++i) {
      const int kk = half * 8 + (i < 4 ? 2 * i : 16 + 2 * (i - 4));
      pa[2 * i] = Ps[w][laneM][kk];
      pa[2 * i + 1] = Ps[w][laneM][kk + 1];
    }
    // O += P(16x32) * V(32x96)
#pragma unroll
    for (int dt = 0; dt < 6; ++dt) {
      v16h bv;
#pragma unroll
      for (int v = 0; v < 8; ++v) {
        const int kk = half * 8 + (v < 4 ? 2 * v : 16 + 2 * (v - 4));
        bv[2 * v] = Vs[kk][dt * 16 + laneM];
        bv[2 * v + 1] = Vs[kk + 1][dt * 16 + laneM];
      }
      o[dt] = wmma16x16x32(pa, bv, o[dt]);
    }
    __syncthreads();
  }

#pragma unroll
  for (int dt = 0; dt < 6; ++dt)
#pragma unroll
    for (int r = 0; r < 8; ++r) {
      const int row = half * 8 + r;
      attnh[((size_t)(b * LL + qbase + row)) * DD + h * HD + dt * 16 + laneM] =
          (f16)(o[dt][r] / lrun[r]);
    }
}

// ---------------------------------------------------------------------------
// LayerNorm over D=768, one block per row.
// ---------------------------------------------------------------------------
__global__ __launch_bounds__(256) void layernorm_k(
    const float* __restrict__ in, const float* __restrict__ g,
    const float* __restrict__ bta, float* __restrict__ outF,
    f16* __restrict__ outH) {
  __shared__ float s1[256], s2[256];
  const int t = threadIdx.x;
  const size_t row = blockIdx.x;
  const float* p = in + row * DD;
  float vals[3], a = 0.f, q = 0.f;
#pragma unroll
  for (int j = 0; j < 3; ++j) {
    const float v = p[t + j * 256];
    vals[j] = v;
    a += v;
    q += v * v;
  }
  s1[t] = a;
  s2[t] = q;
  __syncthreads();
  for (int st = 128; st > 0; st >>= 1) {
    if (t < st) {
      s1[t] += s1[t + st];
      s2[t] += s2[t + st];
    }
    __syncthreads();
  }
  const float mean = s1[0] * (1.0f / DD);
  const float var = s2[0] * (1.0f / DD) - mean * mean;
  const float inv = rsqrtf(var + 1e-5f);
#pragma unroll
  for (int j = 0; j < 3; ++j) {
    const int c = t + j * 256;
    const float y = (vals[j] - mean) * inv * g[c] + bta[c];
    outF[row * DD + c] = y;
    if (outH) outH[row * DD + c] = (f16)y;
  }
}

extern "C" void kernel_launch(void* const* d_in, const int* in_sizes, int n_in,
                              void* d_out, int out_size, void* d_ws,
                              size_t ws_size, hipStream_t stream) {
  const float* x = (const float*)d_in[0];
  const float* abias = (const float*)d_in[1];
  const float* in_w = (const float*)d_in[2];
  const float* in_b = (const float*)d_in[3];
  const float* out_w = (const float*)d_in[4];
  const float* out_b = (const float*)d_in[5];
  const float* ff1_w = (const float*)d_in[6];
  const float* ff1_b = (const float*)d_in[7];
  const float* ff2_w = (const float*)d_in[8];
  const float* ff2_b = (const float*)d_in[9];
  const float* ln1_g = (const float*)d_in[10];
  const float* ln1_b = (const float*)d_in[11];
  const float* ln2_g = (const float*)d_in[12];
  const float* ln2_b = (const float*)d_in[13];
  float* out = (float*)d_out;

  char* base = (char*)d_ws;
  size_t off = 0;
  auto alloc = [&](size_t bytes) -> void* {
    void* p = base + off;
    off = (off + bytes + 255) & ~(size_t)255;
    return p;
  };
  f16* xh = (f16*)alloc((size_t)NTOK * DD * 2);
  f16* wqkvh = (f16*)alloc((size_t)3 * DD * DD * 2);
  f16* wouth = (f16*)alloc((size_t)DD * DD * 2);
  f16* wff1h = (f16*)alloc((size_t)DFF * DD * 2);
  f16* wff2h = (f16*)alloc((size_t)DD * DFF * 2);
  f16* qkvh = (f16*)alloc((size_t)3 * BBS * HHN * LL * HD * 2);
  f16* atth = (f16*)alloc((size_t)NTOK * DD * 2);
  float* y1 = (float*)alloc((size_t)NTOK * DD * 4);
  float* x1 = (float*)alloc((size_t)NTOK * DD * 4);
  f16* x1h = (f16*)alloc((size_t)NTOK * DD * 2);
  f16* hbuf = (f16*)alloc((size_t)NTOK * DFF * 2);
  float* y2 = (float*)alloc((size_t)NTOK * DD * 4);
  (void)in_sizes; (void)n_in; (void)out_size; (void)ws_size;

  cvt_f32_f16<<<1024, 256, 0, stream>>>(x, xh, NTOK * DD);
  cvt_f32_f16<<<1024, 256, 0, stream>>>(in_w, wqkvh, 3 * DD * DD);
  cvt_f32_f16<<<512, 256, 0, stream>>>(out_w, wouth, DD * DD);
  cvt_f32_f16<<<1024, 256, 0, stream>>>(ff1_w, wff1h, DFF * DD);
  cvt_f32_f16<<<1024, 256, 0, stream>>>(ff2_w, wff2h, DD * DFF);

  // QKV projection, scatter to head-major q/k/v (f16)
  gemm_wmma<0><<<dim3(2304 / 128, NTOK / 128), 256, 0, stream>>>(
      xh, DD, wqkvh, in_b, 2304, DD, nullptr, qkvh, nullptr);
  // fused attention (scores + bias + softmax + P*V), no score materialization
  flash_attn<<<dim3(LL / 128, BBS * HHN), 256, 0, stream>>>(qkvh, abias, atth);
  // output projection + bias + residual(x) -> f32
  gemm_wmma<1><<<dim3(DD / 128, NTOK / 128), 256, 0, stream>>>(
      atth, DD, wouth, out_b, DD, DD, y1, nullptr, x);
  layernorm_k<<<NTOK, 256, 0, stream>>>(y1, ln1_g, ln1_b, x1, x1h);
  // FF1 + exact GELU -> f16
  gemm_wmma<2><<<dim3(DFF / 128, NTOK / 128), 256, 0, stream>>>(
      x1h, DD, wff1h, ff1_b, DFF, DD, nullptr, hbuf, nullptr);
  // FF2 + bias + residual(x1) -> f32
  gemm_wmma<1><<<dim3(DD / 128, NTOK / 128), 256, 0, stream>>>(
      hbuf, DFF, wff2h, ff2_b, DD, DFF, y2, nullptr, x1);
  layernorm_k<<<NTOK, 256, 0, stream>>>(y2, ln2_g, ln2_b, out, nullptr);
}
